// GNMT_22110491640160
// MI455X (gfx1250) — matrix operand order repaired and checked by reading, not compiled, not run on hardware
//
#include <hip/hip_runtime.h>

typedef _Float16 h16_t;
typedef _Float16 v16h __attribute__((ext_vector_type(16)));
typedef _Float16 h8   __attribute__((ext_vector_type(8)));
typedef float    v8f  __attribute__((ext_vector_type(8)));

static __device__ __forceinline__ float sigmoidf_(float x) {
    return 1.0f / (1.0f + __expf(-x));
}

// ---------------------------------------------------------------------------
// WMMA GEMM: C[M,N] = A[M,K](f16) @ W[N,K](f16)^T (+ bias[N]) (+= C if accum)
// One wave computes one 16x16 D tile with V_WMMA_F32_16X16X32_F16.
// A fragment (16x32, 16-bit, ISA 7.12.2): lane l holds row (l&15);
//   k-chunks [koff,koff+8) and [16+koff,16+koff+8), koff = (l>>4)*8.
// B fragment (32x16, 16-bit): lane l holds col (l&15);
//   k-chunks [kb,kb+8) and [16+kb,16+kb+8), kb = (l>>4)*8.
// C/D (16x16 f32): VGPR v, lane l -> row = v + (l>>4)*8, col = l&15.
// Requires M%16==0, N%64==0 (4 waves/block along N), K%32==0; EXEC all ones.
// ---------------------------------------------------------------------------
__global__ void gemm_f16_wmma_k(const h16_t* __restrict__ A, int lda,
                                const h16_t* __restrict__ W, int ldw,
                                const float* __restrict__ bias,
                                float* __restrict__ C, long ldc,
                                int K, int accum)
{
    const int lane  = threadIdx.x;                       // 0..31
    const int ntile = blockIdx.x * blockDim.y + threadIdx.y;
    const int mtile = blockIdx.y;

    const int r  = lane & 15;
    const int ak = (lane >> 4) << 3;                     // 0 or 8
    const int bk = (lane >> 4) << 3;                     // 0 or 8

    const h16_t* Ar = A + (size_t)(mtile * 16 + r) * lda;
    const h16_t* Wr = W + (size_t)(ntile * 16 + r) * ldw;

    const int crow = (lane >> 4) * 8;
    const int ccol = lane & 15;
    float* Cr = C + (size_t)(mtile * 16 + crow) * ldc + (size_t)ntile * 16 + ccol;

    v8f acc;
    if (accum) {
#pragma unroll
        for (int v = 0; v < 8; ++v) acc[v] = Cr[(size_t)v * ldc];
    } else {
#pragma unroll
        for (int v = 0; v < 8; ++v) acc[v] = 0.0f;
    }

    for (int k = 0; k < K; k += 32) {
        h8 a0 = *(const h8*)(Ar + k + ak);
        h8 a1 = *(const h8*)(Ar + k + 16 + ak);
        h8 b0 = *(const h8*)(Wr + k + bk);
        h8 b1 = *(const h8*)(Wr + k + 16 + bk);
        v16h av, bv;
#pragma unroll
        for (int i = 0; i < 8; ++i) {
            av[i] = a0[i]; av[i + 8] = a1[i];
            bv[i] = b0[i]; bv[i + 8] = b1[i];
        }
        acc = __builtin_amdgcn_wmma_f32_16x16x32_f16(
            /*neg_a=*/false, av, /*neg_b=*/false, bv,
            /*c_mod=*/(short)0, acc, /*reuse_a=*/false, /*reuse_b=*/false);
    }

    const float bb = bias ? bias[(size_t)ntile * 16 + ccol] : 0.0f;
#pragma unroll
    for (int v = 0; v < 8; ++v) Cr[(size_t)v * ldc] = acc[v] + bb;
}

// ---------------------------------------------------------------------------
// LSTM gate elementwise update. gates rows hold [i|f|g|o] (4H). Optional
// validity masking by src length (encoder), optional residual added into the
// new hidden state (decoder layers >= 2). Writes f32 state + f16 copy.
// ---------------------------------------------------------------------------
__global__ void lstm_elem(const float* __restrict__ gates, long gstride,
                          float* __restrict__ h, float* __restrict__ c,
                          h16_t* __restrict__ h16,
                          float* __restrict__ out_seq, long out_stride,
                          const float* __restrict__ resid,
                          const int* __restrict__ lengths, int t,
                          int H, int n)
{
    int idx = blockIdx.x * blockDim.x + threadIdx.x;
    if (idx >= n) return;
    int b  = idx / H;
    int hh = idx - b * H;
    const float* g = gates + (size_t)b * gstride;
    float gi = g[hh];
    float gf = g[H + hh];
    float gg = g[2 * H + hh];
    float go = g[3 * H + hh];
    float cc = c[idx];
    float c2 = sigmoidf_(gf) * cc + sigmoidf_(gi) * tanhf(gg);
    float h2 = sigmoidf_(go) * tanhf(c2);
    if (resid) h2 += resid[idx];
    bool m = lengths ? (t < lengths[b]) : true;
    float hn = m ? h2 : h[idx];
    float cn = m ? c2 : cc;
    h[idx] = hn;
    c[idx] = cn;
    h16[idx] = (h16_t)hn;
    if (out_seq) out_seq[(size_t)b * out_stride + hh] = m ? h2 : 0.0f;
}

__global__ void gather_embed(const int* __restrict__ tok,
                             const float* __restrict__ tab,
                             h16_t* __restrict__ out, int E, long n)
{
    long idx = (long)blockIdx.x * blockDim.x + threadIdx.x;
    if (idx >= n) return;
    long r = idx / E;
    int  e = (int)(idx - r * E);
    int  tk = tok[r];
    float v = (tk == 0) ? 0.0f : tab[(size_t)tk * E + e];   // table.at[0].set(0)
    out[idx] = (h16_t)v;
}

__global__ void cvt_f32_h16(const float* __restrict__ in, h16_t* __restrict__ out, long n)
{
    long idx = (long)blockIdx.x * blockDim.x + threadIdx.x;
    if (idx >= n) return;
    out[idx] = (h16_t)in[idx];
}

__global__ void add2_cvt(const float* __restrict__ a, const float* __restrict__ b,
                         float* __restrict__ o, h16_t* __restrict__ o16, long n)
{
    long idx = (long)blockIdx.x * blockDim.x + threadIdx.x;
    if (idx >= n) return;
    float s = a[idx] + b[idx];
    o[idx] = s;
    if (o16) o16[idx] = (h16_t)s;
}

__global__ void acc_cvt(float* __restrict__ o, const float* __restrict__ r,
                        h16_t* __restrict__ o16, long n)
{
    long idx = (long)blockIdx.x * blockDim.x + threadIdx.x;
    if (idx >= n) return;
    float s = o[idx] + r[idx];
    o[idx] = s;
    o16[idx] = (h16_t)s;
}

__global__ void concat2_f32(const float* __restrict__ A, long sA, int WA,
                            const float* __restrict__ B, long sB, int WB,
                            h16_t* __restrict__ dst, long n)
{
    long idx = (long)blockIdx.x * blockDim.x + threadIdx.x;
    if (idx >= n) return;
    int  W = WA + WB;
    long r = idx / W;
    int  c = (int)(idx - r * W);
    dst[idx] = (h16_t)((c < WA) ? A[r * sA + c] : B[r * sB + (c - WA)]);
}

__global__ void concat2_h16(const h16_t* __restrict__ A, long sA, int WA,
                            const h16_t* __restrict__ B, long sB, int WB,
                            h16_t* __restrict__ dst, long n)
{
    long idx = (long)blockIdx.x * blockDim.x + threadIdx.x;
    if (idx >= n) return;
    int  W = WA + WB;
    long r = idx / W;
    int  c = (int)(idx - r * W);
    dst[idx] = (c < WA) ? A[r * sA + c] : B[r * sB + (c - WA)];
}

// One block per batch row: scores -> masked softmax -> context + attn weights.
__global__ void attention_k(const float* __restrict__ enc_out,   // B x Ls x H
                            const float* __restrict__ q,         // B x H
                            const int*   __restrict__ lengths,
                            h16_t* __restrict__ ctx16,           // B x H
                            float* __restrict__ attn_out, long attn_rstride,
                            int Ls, int H)
{
    int b = blockIdx.x;
    __shared__ float aw[64];
    const float* qb = q + (size_t)b * H;

    for (int s = threadIdx.x; s < Ls; s += blockDim.x) {
        const float* e = enc_out + ((size_t)b * Ls + s) * H;
        float d = 0.0f;
        for (int hh = 0; hh < H; ++hh) d += e[hh] * qb[hh];
        aw[s] = d;
    }
    __syncthreads();
    if (threadIdx.x == 0) {
        int len = lengths[b];
        float mx = -3.4e38f;
        for (int s = 0; s < len; ++s) mx = fmaxf(mx, aw[s]);
        float sum = 0.0f;
        for (int s = 0; s < Ls; ++s) {
            float e = (s < len) ? __expf(aw[s] - mx) : 0.0f;  // pad -> -inf -> 0
            aw[s] = e;
            sum += e;
        }
        float inv = 1.0f / sum;
        for (int s = 0; s < Ls; ++s) aw[s] *= inv;
    }
    __syncthreads();
    for (int s = threadIdx.x; s < Ls; s += blockDim.x)
        attn_out[(size_t)b * attn_rstride + s] = aw[s];
    for (int hh = threadIdx.x; hh < H; hh += blockDim.x) {
        float d = 0.0f;
        for (int s = 0; s < Ls; ++s)
            d += aw[s] * enc_out[((size_t)b * Ls + s) * H + hh];
        ctx16[(size_t)b * H + hh] = (h16_t)d;
    }
}

// ---------------------------------------------------------------------------
extern "C" void kernel_launch(void* const* d_in, const int* in_sizes, int n_in,
                              void* d_out, int out_size, void* d_ws, size_t ws_size,
                              hipStream_t stream)
{
    (void)in_sizes; (void)n_in; (void)out_size; (void)ws_size;
    constexpr int B = 32, Ls = 64, Lt = 64, E = 512, H = 512;
    constexpr int G = 4 * H;        // 2048
    constexpr int V = 32000;
    constexpr int K2 = 1024;        // E+H == 2H
    constexpr int NL = 4;

    const int*   src_tokens = (const int*)  d_in[0];
    const int*   src_len    = (const int*)  d_in[1];
    const int*   tgt_tokens = (const int*)  d_in[2];
    const float* enc_embed  = (const float*)d_in[3];
    const float* el0_wih    = (const float*)d_in[4];
    const float* el0_whh    = (const float*)d_in[5];
    const float* el0_b      = (const float*)d_in[6];
    const float* proj_w     = (const float*)d_in[7];
    const float* proj_b     = (const float*)d_in[8];
    const float* enc_wih    = (const float*)d_in[9];
    const float* enc_whh    = (const float*)d_in[10];
    const float* enc_b      = (const float*)d_in[11];
    const float* dec_embed  = (const float*)d_in[12];
    const float* d0_wih     = (const float*)d_in[13];
    const float* d0_whh     = (const float*)d_in[14];
    const float* d0_b       = (const float*)d_in[15];
    const float* dc_wih     = (const float*)d_in[16];
    const float* dc_whh     = (const float*)d_in[17];
    const float* dc_b       = (const float*)d_in[18];
    const float* attn_w     = (const float*)d_in[19];
    const float* out_w      = (const float*)d_in[20];
    const float* out_b      = (const float*)d_in[21];

    float* out_logits = (float*)d_out;
    float* out_attn   = out_logits + (size_t)B * Lt * V;

    // ---- bump allocator over workspace ----
    char* p = (char*)d_ws;
    auto bump  = [&](size_t bytes) -> char* {
        char* q = p;
        p += (bytes + 255) & ~(size_t)255;
        return q;
    };
    auto bumpH = [&](size_t n) -> h16_t* { return (h16_t*)bump(n * sizeof(h16_t)); };
    auto bumpF = [&](size_t n) -> float* { return (float*)bump(n * sizeof(float)); };

    // f16 weight copies (out_w f16 = 65 MB -> resident in 192 MB L2)
    h16_t* w_el0_wih = bumpH((size_t)2 * G * E);
    h16_t* w_el0_whh = bumpH((size_t)2 * G * H);
    h16_t* w_proj    = bumpH((size_t)H * 2 * H);
    h16_t* w_enc_wih = bumpH((size_t)3 * G * H);
    h16_t* w_enc_whh = bumpH((size_t)3 * G * H);
    h16_t* w_d0_wih  = bumpH((size_t)G * K2);
    h16_t* w_d0_whh  = bumpH((size_t)G * H);
    h16_t* w_dec_wih = bumpH((size_t)3 * G * H);
    h16_t* w_dec_whh = bumpH((size_t)3 * G * H);
    h16_t* w_attn    = bumpH((size_t)H * H);
    h16_t* w_out     = bumpH((size_t)V * 2 * H);

    h16_t* emb16  = bumpH((size_t)B * Ls * E);
    h16_t* demb16 = bumpH((size_t)B * Lt * E);
    h16_t* x16    = bumpH((size_t)B * Ls * H);
    h16_t* xcat16 = bumpH((size_t)B * Ls * 2 * H);
    h16_t* li16   = bumpH((size_t)B * K2);
    h16_t* cat16  = bumpH((size_t)B * K2);
    h16_t* ctx16  = bumpH((size_t)B * H);
    h16_t* tmp16  = bumpH((size_t)B * H);
    h16_t* hS16[NL];
    for (int i = 0; i < NL; ++i) hS16[i] = bumpH((size_t)B * H);

    float* gx   = bumpF((size_t)B * Ls * G);
    float* of   = bumpF((size_t)B * Ls * H);
    float* ob   = bumpF((size_t)B * Ls * H);
    float* xbuf = bumpF((size_t)B * Ls * H);
    float* obuf = bumpF((size_t)B * Ls * H);
    float* hf = bumpF((size_t)B * H);
    float* cf = bumpF((size_t)B * H);
    float* hb = bumpF((size_t)B * H);
    float* cb = bumpF((size_t)B * H);
    float* hS[NL];
    float* cS[NL];
    for (int i = 0; i < NL; ++i) { hS[i] = bumpF((size_t)B * H); cS[i] = bumpF((size_t)B * H); }
    float* dg = bumpF((size_t)B * G);
    float* qv = bumpF((size_t)B * H);

    // ---- helpers ----
    auto cvt = [&](const float* in, h16_t* out, long n) {
        cvt_f32_h16<<<dim3((unsigned)((n + 255) / 256)), dim3(256), 0, stream>>>(in, out, n);
    };
    auto gemm = [&](const h16_t* A, int lda, const h16_t* W, int ldw,
                    const float* bias, float* C, long ldc,
                    int M, int N, int K, int accum) {
        gemm_f16_wmma_k<<<dim3(N / 64, M / 16), dim3(32, 4), 0, stream>>>(
            A, lda, W, ldw, bias, C, ldc, K, accum);
    };
    auto lstm = [&](const float* gates, long gstride, float* h, float* c, h16_t* h16v,
                    float* oseq, long ostride, const float* resid,
                    const int* len, int t) {
        int n = B * H;
        lstm_elem<<<dim3((n + 255) / 256), dim3(256), 0, stream>>>(
            gates, gstride, h, c, h16v, oseq, ostride, resid, len, t, H, n);
    };
    auto zero = [&](void* ptr, size_t bytes) { hipMemsetAsync(ptr, 0, bytes, stream); };

    // ---- weight conversion (once per launch; ~210 MB read @23.3 TB/s ≈ 9 µs) ----
    cvt(el0_wih, w_el0_wih, (long)2 * G * E);
    cvt(el0_whh, w_el0_whh, (long)2 * G * H);
    cvt(proj_w,  w_proj,    (long)H * 2 * H);
    cvt(enc_wih, w_enc_wih, (long)3 * G * H);
    cvt(enc_whh, w_enc_whh, (long)3 * G * H);
    cvt(d0_wih,  w_d0_wih,  (long)G * K2);
    cvt(d0_whh,  w_d0_whh,  (long)G * H);
    cvt(dc_wih,  w_dec_wih, (long)3 * G * H);
    cvt(dc_whh,  w_dec_whh, (long)3 * G * H);
    cvt(attn_w,  w_attn,    (long)H * H);
    cvt(out_w,   w_out,     (long)V * 2 * H);

    // ---- embeddings ----
    {
        long n = (long)B * Ls * E;
        gather_embed<<<dim3((unsigned)((n + 255) / 256)), dim3(256), 0, stream>>>(
            src_tokens, enc_embed, emb16, E, n);
    }
    {
        long n = (long)B * Lt * E;
        gather_embed<<<dim3((unsigned)((n + 255) / 256)), dim3(256), 0, stream>>>(
            tgt_tokens, dec_embed, demb16, E, n);
    }

    // ---- encoder layer 0, forward direction ----
    gemm(emb16, E, w_el0_wih, E, el0_b, gx, G, B * Ls, G, E, 0);   // x@Wih^T+b, all t
    zero(hf, (size_t)B * H * 4); zero(cf, (size_t)B * H * 4); zero(tmp16, (size_t)B * H * 2);
    for (int t = 0; t < Ls; ++t) {
        gemm(tmp16, H, w_el0_whh, H, nullptr, gx + (size_t)t * G, (long)Ls * G, B, G, H, 1);
        lstm(gx + (size_t)t * G, (long)Ls * G, hf, cf, tmp16,
             of + (size_t)t * H, (long)Ls * H, nullptr, src_len, t);
    }

    // ---- encoder layer 0, backward direction ----
    gemm(emb16, E, w_el0_wih + (size_t)G * E, E, el0_b + G, gx, G, B * Ls, G, E, 0);
    zero(hb, (size_t)B * H * 4); zero(cb, (size_t)B * H * 4); zero(tmp16, (size_t)B * H * 2);
    for (int t = Ls - 1; t >= 0; --t) {
        gemm(tmp16, H, w_el0_whh + (size_t)G * H, H, nullptr,
             gx + (size_t)t * G, (long)Ls * G, B, G, H, 1);
        lstm(gx + (size_t)t * G, (long)Ls * G, hb, cb, tmp16,
             ob + (size_t)t * H, (long)Ls * H, nullptr, src_len, t);
    }

    // decoder layer-0 initial state: hf+hb / cf+cb
    {
        long n = (long)B * H;
        add2_cvt<<<dim3((unsigned)((n + 255) / 256)), dim3(256), 0, stream>>>(hf, hb, hS[0], hS16[0], n);
        add2_cvt<<<dim3((unsigned)((n + 255) / 256)), dim3(256), 0, stream>>>(cf, cb, cS[0], (h16_t*)nullptr, n);
    }

    // ---- projection: x = concat(of, ob) @ proj_w^T + proj_b ----
    {
        long n = (long)B * Ls * 2 * H;
        concat2_f32<<<dim3((unsigned)((n + 255) / 256)), dim3(256), 0, stream>>>(
            of, (long)H, H, ob, (long)H, H, xcat16, n);
    }
    gemm(xcat16, 2 * H, w_proj, 2 * H, proj_b, xbuf, H, B * Ls, H, 2 * H, 0);
    cvt(xbuf, x16, (long)B * Ls * H);

    // ---- encoder upper layers (state feeds decoder layer i+1) ----
    for (int i = 0; i < 3; ++i) {
        gemm(x16, H, w_enc_wih + (size_t)i * G * H, H, enc_b + (size_t)i * G,
             gx, G, B * Ls, G, H, 0);
        zero(hS[i + 1], (size_t)B * H * 4);
        zero(cS[i + 1], (size_t)B * H * 4);
        zero(hS16[i + 1], (size_t)B * H * 2);
        for (int t = 0; t < Ls; ++t) {
            gemm(hS16[i + 1], H, w_enc_whh + (size_t)i * G * H, H, nullptr,
                 gx + (size_t)t * G, (long)Ls * G, B, G, H, 1);
            lstm(gx + (size_t)t * G, (long)Ls * G, hS[i + 1], cS[i + 1], hS16[i + 1],
                 obuf + (size_t)t * H, (long)Ls * H, nullptr, src_len, t);
        }
        long n = (long)B * Ls * H;
        if (i >= 1) {   // residual: o = o + res (res == previous x)
            acc_cvt<<<dim3((unsigned)((n + 255) / 256)), dim3(256), 0, stream>>>(obuf, xbuf, x16, n);
        } else {
            cvt(obuf, x16, n);
        }
        float* tsw = xbuf; xbuf = obuf; obuf = tsw;   // x = o
    }
    const float* enc_out = xbuf;   // B x Ls x H

    // ---- decoder ----
    zero(ctx16, (size_t)B * H * 2);
    for (int t = 0; t < Lt; ++t) {
        // li = concat(embed_t, ctx)
        {
            long n = (long)B * K2;
            concat2_h16<<<dim3((unsigned)((n + 255) / 256)), dim3(256), 0, stream>>>(
                demb16 + (size_t)t * E, (long)Lt * E, E, ctx16, (long)H, H, li16, n);
        }
        // layer 0
        gemm(li16, K2, w_d0_wih, K2, d0_b, dg, G, B, G, K2, 0);
        gemm(hS16[0], H, w_d0_whh, H, nullptr, dg, G, B, G, H, 1);
        lstm(dg, G, hS[0], cS[0], hS16[0], nullptr, 0, nullptr, nullptr, 0);
        // layers 1..3 (residual into state for layers >= 2)
        for (int i = 0; i < 3; ++i) {
            gemm(hS16[i], H, w_dec_wih + (size_t)i * G * H, H, dc_b + (size_t)i * G,
                 dg, G, B, G, H, 0);
            gemm(hS16[i + 1], H, w_dec_whh + (size_t)i * G * H, H, nullptr, dg, G, B, G, H, 1);
            lstm(dg, G, hS[i + 1], cS[i + 1], hS16[i + 1], nullptr, 0,
                 (i >= 1) ? hS[i] : nullptr, nullptr, 0);
        }
        // attention: q = top @ attn_w^T ; scores/softmax/ctx
        gemm(hS16[3], H, w_attn, H, nullptr, qv, H, B, H, H, 0);
        attention_k<<<dim3(B), dim3(256), 0, stream>>>(
            enc_out, qv, src_len, ctx16, out_attn + (size_t)t * Ls, (long)Lt * Ls, Ls, H);
        // logits = concat(top, ctx) @ out_w^T + out_b  -> straight into d_out
        {
            long n = (long)B * K2;
            concat2_h16<<<dim3((unsigned)((n + 255) / 256)), dim3(256), 0, stream>>>(
                hS16[3], (long)H, H, ctx16, (long)H, H, cat16, n);
        }
        gemm(cat16, K2, w_out, 2 * H, out_b, out_logits + (size_t)t * V, (long)Lt * V,
             B, V, K2, 0);
    }
}
